// SentenceEncoder_386547057264
// MI455X (gfx1250) — matrix operand order repaired and checked by reading, not compiled, not run on hardware
//
#include <hip/hip_runtime.h>

// Sizes from the reference
#define B_   256
#define T_   128
#define D_   512     // WORD_DIM
#define H_   1024    // HIDDEN
#define K4_  1536    // D_ + H_  (fused [x_t, h] contraction dim)
#define G4_  4096    // 4*H_     (gate dim)
#define ALD  1544    // A-tile LDS row stride (1536 + 8 pad -> stride%64dw == 4)

typedef __bf16 bf16;
typedef __attribute__((ext_vector_type(4)))  __bf16 v4bf;
typedef __attribute__((ext_vector_type(8)))  __bf16 v8bf;
typedef __attribute__((ext_vector_type(16))) __bf16 v16bf;
typedef __attribute__((ext_vector_type(8)))  float  v8f;
typedef __attribute__((ext_vector_type(4)))  float  v4f;

__device__ __forceinline__ float sigf(float x)     { return 1.f / (1.f + __expf(-x)); }
__device__ __forceinline__ float tanhfast(float x) { return 1.f - 2.f / (__expf(2.f * x) + 1.f); }

// ---------------------------------------------------------------------------
// One-time prep: fuse W_ih|W_hh into bf16 W[4096][1536], bias = b_ih + b_hh
// ---------------------------------------------------------------------------
__global__ __launch_bounds__(256) void prep_weights(const float* __restrict__ Wih,
                                                    const float* __restrict__ Whh,
                                                    const float* __restrict__ bih,
                                                    const float* __restrict__ bhh,
                                                    bf16* __restrict__ Wf,
                                                    float* __restrict__ bias)
{
  long gid = (long)blockIdx.x * 256 + threadIdx.x;   // over 4096*1536/4
  long e = gid * 4;
  int j = (int)(e / K4_);
  int k = (int)(e % K4_);
  v4f v = (k < D_) ? *(const v4f*)(Wih + (long)j * D_ + k)
                   : *(const v4f*)(Whh + (long)j * H_ + (k - D_));
  v4bf o = { (bf16)v[0], (bf16)v[1], (bf16)v[2], (bf16)v[3] };
  *(v4bf*)(Wf + e) = o;
  if (gid < G4_) bias[gid] = bih[gid] + bhh[gid];
}

// ---------------------------------------------------------------------------
// Embedding gather + fp32 -> bf16 for one sequence: Xe[B][T][D]
// ---------------------------------------------------------------------------
__global__ __launch_bounds__(256) void gather_embed(const int* __restrict__ tok,
                                                    const float* __restrict__ emb,
                                                    bf16* __restrict__ Xe)
{
  long gid = (long)blockIdx.x * 256 + threadIdx.x;   // over B*T*D/4
  long e = gid * 4;
  int  d  = (int)(e & (D_ - 1));
  long bt = e >> 9;                                  // b*T + t
  int tokid = tok[bt];
  v4f v = *(const v4f*)(emb + (long)tokid * D_ + d);
  v4bf o = { (bf16)v[0], (bf16)v[1], (bf16)v[2], (bf16)v[3] };
  *(v4bf*)(Xe + e) = o;
}

// ---------------------------------------------------------------------------
// Per-sequence state init: h0 = 0, c = 0, hmax = -inf
// ---------------------------------------------------------------------------
__global__ __launch_bounds__(256) void init_state(bf16* __restrict__ h0,
                                                  float* __restrict__ c,
                                                  float* __restrict__ hmax)
{
  int i = blockIdx.x * 256 + threadIdx.x;            // over B*H
  h0[i]   = (bf16)0.f;
  c[i]    = 0.f;
  hmax[i] = -3.4e38f;
}

// ---------------------------------------------------------------------------
// One LSTM timestep, fully fused:
//   A = [x_t, h_{t-1}] staged once per block into LDS (48 KB, padded rows)
//   gates = A @ Wf^T  (bf16 WMMA, fp32 acc; B streamed from L2-resident Wf)
//   gates -> LDS -> elementwise LSTM update -> c, h_out(bf16), hmax
// Block = 16 batch rows x 64 hidden cols (owns all 4 gate slices for them).
// 8 waves: wave w -> gate g=w>>1, col-pair (w&1); 2 accumulators per wave.
// ---------------------------------------------------------------------------
__global__ __launch_bounds__(256) void lstm_step(const bf16* __restrict__ Xe, int t,
                                                 const bf16* __restrict__ Wf,
                                                 const float* __restrict__ bias,
                                                 const bf16* __restrict__ h_in,
                                                 bf16* __restrict__ h_out,
                                                 float* __restrict__ c,
                                                 float* __restrict__ hmax)
{
  __shared__ __align__(16) bf16 A_lds[16][ALD];      // 49,408 B: fused A tile
  __shared__ __align__(16) float lds_g[4][16][64];   // 16 KB: gates for this block

  const int tid  = threadIdx.x;
  const int lane = tid & 31;
  const int w    = tid >> 5;
  const int bt   = blockIdx.x >> 4;                  // batch tile (16 rows)
  const int ct   = blockIdx.x & 15;                  // hidden-col tile (64 cols)

  // ---- cooperative fill of A_lds: rows = batch, cols = [x_t(512) | h(1024)]
  {
    const int trow = tid >> 4;                       // 0..15 batch row
    const int tcol = tid & 15;                       // 16 threads per row
    const long xbase = ((long)(bt * 16 + trow) * T_ + t) * D_;
    const long hbase = (long)(bt * 16 + trow) * H_;
    #pragma unroll
    for (int cnk = 0; cnk < 12; ++cnk) {             // 12 x 8 elems = 96/thread
      const int k8 = tcol * 8 + cnk * 128;           // cnk<4 -> x region (k8<512)
      v8bf v = (k8 < D_) ? *(const v8bf*)(Xe  + xbase + k8)
                         : *(const v8bf*)(h_in + hbase + (k8 - D_));
      *(v8bf*)(&A_lds[trow][k8]) = v;
    }
  }
  __syncthreads();

  const int g     = w >> 1;                          // gate 0..3 (i,f,g,o)
  const int npair = w & 1;
  const int n0    = g * H_ + ct * 64 + npair * 32;   // global gate column of tile 0
  const int lm    = lane & 15;
  const int lh    = lane >> 4;

  // A-fragment (16x32 bf16 ISA layout): row = lane%16,
  // per-lane K chunks {kA..kA+7, kA+16..kA+23}, kA = (lane/16)*8  -> from LDS
  const int kA = lh * 8;

  // B-fragment (32x16 bf16): col = lane%16, same K chunks as A:
  // {kB..kB+7, kB+16..kB+23}, kB = (lane/16)*8
  const long bcol0 = (long)(n0 + lm)      * K4_;
  const long bcol1 = (long)(n0 + 16 + lm) * K4_;
  const int  kB    = lh * 8;

  v8f acc0 = {}; v8f acc1 = {};

  #pragma unroll 4
  for (int kk = 0; kk < 48; ++kk) {                  // K = 1536 = 48 * 32
    const int k0 = kk * 32;
    const bf16* pa = &A_lds[lm][k0 + kA];
    v8bf a_lo = *(const v8bf*)pa;                    // ds_load_b128
    v8bf a_hi = *(const v8bf*)(pa + 16);             // ds_load_b128
    v16bf afrag = __builtin_shufflevector(a_lo, a_hi,
        0, 1, 2, 3, 4, 5, 6, 7, 8, 9, 10, 11, 12, 13, 14, 15);
    v16bf b0 = __builtin_shufflevector(*(const v8bf*)(Wf + bcol0 + k0 + kB), *(const v8bf*)(Wf + bcol0 + k0 + kB + 16),
        0, 1, 2, 3, 4, 5, 6, 7, 8, 9, 10, 11, 12, 13, 14, 15);
    v16bf b1 = __builtin_shufflevector(*(const v8bf*)(Wf + bcol1 + k0 + kB), *(const v8bf*)(Wf + bcol1 + k0 + kB + 16),
        0, 1, 2, 3, 4, 5, 6, 7, 8, 9, 10, 11, 12, 13, 14, 15);
    acc0 = __builtin_amdgcn_wmma_f32_16x16x32_bf16(false, afrag, false, b0,
                                                   (short)0, acc0, false, false);
    acc1 = __builtin_amdgcn_wmma_f32_16x16x32_bf16(false, afrag, false, b1,
                                                   (short)0, acc1, false, false);
  }

  // C/D layout: VGPR r holds row r + 8*(lane>=16), col = lane%16
  const int l0 = npair * 32 + lm;
  #pragma unroll
  for (int r = 0; r < 8; ++r) {
    const int m = r + (lh ? 8 : 0);
    lds_g[g][m][l0]      = acc0[r];
    lds_g[g][m][l0 + 16] = acc1[r];
  }
  __syncthreads();

  // ---- elementwise LSTM update: each thread owns 4 contiguous cols of a row
  const int r    = tid >> 4;                         // 0..15 local batch row
  const int cb   = (tid & 15) * 4;                   // 0..60 local col base
  const int b    = bt * 16 + r;
  const int hidx = ct * 64 + cb;
  const long gi  = (long)b * H_ + hidx;
  if (t + 1 < T_)                                    // warm L2 for next step's x
    __builtin_prefetch(Xe + ((long)b * T_ + (t + 1)) * D_, 0, 0);

  v4f iv = *(const v4f*)&lds_g[0][r][cb];
  v4f fv = *(const v4f*)&lds_g[1][r][cb];
  v4f gv = *(const v4f*)&lds_g[2][r][cb];
  v4f ov = *(const v4f*)&lds_g[3][r][cb];
  v4f bi = *(const v4f*)(bias + hidx);
  v4f bfv = *(const v4f*)(bias + H_     + hidx);
  v4f bg = *(const v4f*)(bias + 2 * H_ + hidx);
  v4f bo = *(const v4f*)(bias + 3 * H_ + hidx);
  v4f cold = *(const v4f*)(c + gi);
  v4f hm   = *(const v4f*)(hmax + gi);
  v4f cn, hmn; v4bf hb;
  #pragma unroll
  for (int q = 0; q < 4; ++q) {
    float cv = sigf(fv[q] + bfv[q]) * cold[q] + sigf(iv[q] + bi[q]) * tanhfast(gv[q] + bg[q]);
    float hv = sigf(ov[q] + bo[q]) * tanhfast(cv);
    cn[q]  = cv;
    hb[q]  = (bf16)hv;
    hmn[q] = fmaxf(hm[q], hv);
  }
  *(v4f*)(c + gi)      = cn;
  *(v4bf*)(h_out + gi) = hb;
  *(v4f*)(hmax + gi)   = hmn;
}

// ---------------------------------------------------------------------------
// out[b][k] = sigmoid([hq, hs] @ W_last^T + b_last)   (512 outputs, trivial)
// ---------------------------------------------------------------------------
__global__ __launch_bounds__(256) void final_linear(const float* __restrict__ hq,
                                                    const float* __restrict__ hs,
                                                    const float* __restrict__ Wl,
                                                    const float* __restrict__ bl,
                                                    float* __restrict__ out)
{
  int tid = blockIdx.x * 256 + threadIdx.x;
  if (tid >= B_ * 2) return;
  int b = tid >> 1, k = tid & 1;
  float acc = bl[k];
  const float* wq  = Wl + (long)k * 2 * H_;
  const float* wsp = wq + H_;
  for (int j = 0; j < H_; ++j)
    acc += hq[(long)b * H_ + j] * wq[j] + hs[(long)b * H_ + j] * wsp[j];
  out[tid] = sigf(acc);
}

// ---------------------------------------------------------------------------
extern "C" void kernel_launch(void* const* d_in, const int* in_sizes, int n_in,
                              void* d_out, int out_size, void* d_ws, size_t ws_size,
                              hipStream_t stream)
{
  (void)in_sizes; (void)n_in; (void)out_size; (void)ws_size;
  const int*   question = (const int*)  d_in[0];
  const int*   sentence = (const int*)  d_in[1];
  const float* emb      = (const float*)d_in[2];
  const float* Wih      = (const float*)d_in[3];
  const float* Whh      = (const float*)d_in[4];
  const float* bih      = (const float*)d_in[5];
  const float* bhh      = (const float*)d_in[6];
  const float* Wl       = (const float*)d_in[7];
  const float* bl       = (const float*)d_in[8];
  float* out = (float*)d_out;

  // Workspace carve-up (~48 MB, all 256B-aligned offsets)
  char*  ws    = (char*)d_ws;
  bf16*  Wf    = (bf16*) (ws + 0);          // 4096*1536*2  = 12,582,912
  float* bias  = (float*)(ws + 12582912);   // 4096*4       =     16,384
  bf16*  Xe    = (bf16*) (ws + 12599296);   // 256*128*512*2= 33,554,432
  bf16*  hA    = (bf16*) (ws + 46153728);   // 256*1024*2   =    524,288
  bf16*  hB    = (bf16*) (ws + 46678016);   // 256*1024*2   =    524,288
  float* cbuf  = (float*)(ws + 47202304);   // 256*1024*4   =  1,048,576
  float* hmaxq = (float*)(ws + 48250880);   // 256*1024*4   =  1,048,576
  float* hmaxs = (float*)(ws + 49299456);   // 256*1024*4   =  1,048,576

  prep_weights<<<(G4_ * K4_) / (4 * 256), 256, 0, stream>>>(Wih, Whh, bih, bhh, Wf, bias);

  const int*   toks[2]  = { question, sentence };
  float*       hmaxp[2] = { hmaxq, hmaxs };
  for (int s = 0; s < 2; ++s) {
    gather_embed<<<(B_ * T_ * D_) / (4 * 256), 256, 0, stream>>>(toks[s], emb, Xe);
    init_state<<<(B_ * H_) / 256, 256, 0, stream>>>(hA, cbuf, hmaxp[s]);
    for (int t = 0; t < T_; ++t) {
      bf16* hin  = (t & 1) ? hB : hA;
      bf16* hout = (t & 1) ? hA : hB;
      lstm_step<<<256, 256, 0, stream>>>(Xe, t, Wf, bias, hin, hout, cbuf, hmaxp[s]);
    }
  }
  final_linear<<<2, 256, 0, stream>>>(hmaxq, hmaxs, Wl, bl, out);
}
